// DragonnetCausalRAG_78520592105867
// MI455X (gfx1250) — matrix operand order repaired and checked by reading, not compiled, not run on hardware
//
#include <hip/hip_runtime.h>
#include <hip/hip_bf16.h>
#include <math.h>

// ---------------- CDNA5 WMMA types ----------------
typedef __bf16 bf16_t;
typedef bf16_t v16bf __attribute__((ext_vector_type(16)));
typedef float  v8f   __attribute__((ext_vector_type(8)));

union BF16x16 { unsigned u[8]; v16bf v; };

__device__ __forceinline__ unsigned pk_bf16(float a, float b) {
  unsigned ua = __float_as_uint(a), ub = __float_as_uint(b);
  ua = (ua + 0x7FFFu + ((ua >> 16) & 1u)) >> 16;   // round-to-nearest-even
  ub = (ub + 0x7FFFu + ((ub >> 16) & 1u)) >> 16;
  return (ub << 16) | (ua & 0xFFFFu);
}

__device__ __forceinline__ v8f wmma_bf16(const unsigned a[8], const unsigned b[8], v8f c) {
  BF16x16 A, B;
#pragma unroll
  for (int i = 0; i < 8; ++i) { A.u[i] = a[i]; B.u[i] = b[i]; }
  return __builtin_amdgcn_wmma_f32_16x16x32_bf16(false, A.v, false, B.v,
                                                 (short)0, c, false, false);
}

__device__ __forceinline__ void top8_insert(float* S, int* I, float v, int idx) {
  if (v > S[7]) {
    float cs = v; int ci = idx;
#pragma unroll
    for (int p = 0; p < 8; ++p) {
      if (cs > S[p]) { float ts = S[p]; int ti = I[p]; S[p] = cs; I[p] = ci; cs = ts; ci = ti; }
    }
  }
}

// ---------------- problem constants ----------------
#define BSZ     512
#define EMBD    256
#define NCORP   500000
#define NTILES  31250      // 500000/16
#define TPC     64         // n-tiles per chunk (1024 columns)
#define NCHUNK  489        // ceil(31250/64)
#define KS_PE   8          // 256/32
#define KS_SIN  68         // 2176/32
#define KS_HID  32         // 1024/32

// workspace byte offsets
#define OFF_PEA   0u                                   // 32*8*256 u32 = 256 KB
#define OFF_CAND  262144u                              // NCHUNK*512*16 float2 = ~32 MB
#define OFF_SIN   (OFF_CAND + 489u*512u*16u*8u)        // 32*68*256 u32
#define OFF_HPRE  (OFF_SIN + 32u*68u*256u*4u)          // 512*1024 f32
#define OFF_HF    (OFF_HPRE + 512u*1024u*4u)           // 32*32*256 u32

// d_out float offsets
#define O_FACT 0
#define O_TARG 512
#define O_CF   1024
#define O_PS   2048
#define O_PL   3072
#define O_SC   4096
#define O_IDX  8192
#define O_S    12288

// ================= Kernel 1: pe = normalize(pf @ Wpe + b) -> bf16 A-frags ==========
__global__ __launch_bounds__(256) void pe_prep_kernel(
    const float* __restrict__ pf, const float* __restrict__ Wpe,
    const float* __restrict__ bpe, unsigned* __restrict__ peA) {
  __shared__ float pfrow[132];
  __shared__ float val[256];
  __shared__ float red[256];
  const int tid = threadIdx.x, b = blockIdx.x;
  if (tid < 130) pfrow[tid] = pf[b * 130 + tid];
  __syncthreads();
  float a = bpe[tid];
  for (int k = 0; k < 130; ++k) a += pfrow[k] * Wpe[k * 256 + tid];
  val[tid] = a;
  red[tid] = a * a;
  __syncthreads();
  for (int off = 128; off > 0; off >>= 1) {
    if (tid < off) red[tid] += red[tid + off];
    __syncthreads();
  }
  float scal = 1.f / fmaxf(sqrtf(red[0]), 1e-12f);
  __syncthreads();
  const int mt = b >> 4, m = b & 15;
  if (tid < 128) {
    int k = 2 * tid, kk = k & 31, ks = k >> 5;
    int h = (kk >> 3) & 1, v = ((kk & 7) >> 1) + 4 * (kk >> 4);
    int l = m + 16 * h;
    peA[(((size_t)mt * KS_PE + ks) * 32 + l) * 8 + v] =
        pk_bf16(val[k] * scal, val[k + 1] * scal);
  }
}

// ======= async DMA of one 16x256 fp32 corpus tile into LDS (ASYNCcnt-tracked) ======
__device__ __forceinline__ void async_fill_tile(const float* __restrict__ corpus,
                                                int t, void* ldsRaw, int tid) {
  const char* g = (const char*)(corpus + (size_t)t * 16 * EMBD) + tid * 16;
  unsigned l = (unsigned)(unsigned long long)ldsRaw + (unsigned)(tid * 16);
#pragma unroll
  for (int j = 0; j < 4; ++j) {
    // per issue: 256 lanes x 16B = 4 KB, memory -> LDS with no VGPR transit
    asm volatile("global_load_async_to_lds_b128 %0, %1, off"
                 :: "v"(l + (unsigned)(j * 4096)), "v"(g + (size_t)j * 4096)
                 : "memory");
  }
}

// ================= Kernel 2: streaming sim GEMM (bf16 WMMA) + fused partial top-8 ==
// Pipeline per tile: async-fill(t+1) overlaps 32 WMMAs of tile t.
__global__ __launch_bounds__(256) void sim_topk_kernel(
    const float* __restrict__ corpus, const unsigned* __restrict__ peA,
    float2* __restrict__ cand) {
  __shared__ float    ldsRaw[16 * 256];   // raw fp32 staging tile (16 KB), async DMA dest
  __shared__ unsigned ldsB[8 * 32 * 8];   // bf16 B-fragments (8 KB)
  __shared__ float    ldsRn[16];          // per-corpus-row squared norms
  __shared__ float    ldsT[8 * 256];      // per-wave 16x16 transpose buffer (8 KB)
  const int tid = threadIdx.x, lane = tid & 31, wid = tid >> 5;
  const int chunk = blockIdx.x;
  const int t0 = chunk * TPC;
  const int t1 = (t0 + TPC < NTILES) ? (t0 + TPC) : NTILES;

  float topS[4][8]; int topI[4][8];
#pragma unroll
  for (int mt = 0; mt < 4; ++mt)
#pragma unroll
    for (int j = 0; j < 8; ++j) { topS[mt][j] = -3.4e38f; topI[mt][j] = 0; }

  const int r  = tid & 15;          // corpus row within tile
  const int kb = (tid >> 4) * 16;   // 16 K-elements per thread

  async_fill_tile(corpus, t0, ldsRaw, tid);   // prologue DMA

  for (int t = t0; t < t1; ++t) {
    asm volatile("s_wait_asynccnt 0x0" ::: "memory");
    __syncthreads();                 // raw(t) in LDS; compute(t-1) fully drained
    if (tid < 16) ldsRn[tid] = 0.f;
    __syncthreads();

    // convert raw fp32 -> bf16 B-fragment layout + row norms (LDS->LDS, no HBM)
    float nrm = 0.f;
#pragma unroll
    for (int j = 0; j < 4; ++j) {
      float4 f = *reinterpret_cast<const float4*>(&ldsRaw[r * 256 + kb + 4 * j]);
      int k = kb + 4 * j;
      int ks = k >> 5, kk = k & 31;
      int l = ((kk >> 3) & 1) * 16 + r;
      int v = ((kk & 7) >> 1) + 4 * (kk >> 4);
      ldsB[(ks * 32 + l) * 8 + v]     = pk_bf16(f.x, f.y);
      ldsB[(ks * 32 + l) * 8 + v + 1] = pk_bf16(f.z, f.w);
      nrm += f.x * f.x + f.y * f.y + f.z * f.z + f.w * f.w;
    }
    atomicAdd(&ldsRn[r], nrm);
    __syncthreads();

    if (t + 1 < t1) async_fill_tile(corpus, t + 1, ldsRaw, tid);  // overlap with WMMA

    v8f acc[4];
#pragma unroll
    for (int mt = 0; mt < 4; ++mt)
#pragma unroll
      for (int i = 0; i < 8; ++i) acc[mt][i] = 0.f;

#pragma unroll
    for (int ks = 0; ks < 8; ++ks) {
      unsigned bf[8];
#pragma unroll
      for (int v = 0; v < 8; ++v) bf[v] = ldsB[(ks * 32 + lane) * 8 + v];
#pragma unroll
      for (int mt = 0; mt < 4; ++mt) {
        const unsigned* ap = &peA[(((size_t)(wid * 4 + mt) * KS_PE + ks) * 32 + lane) * 8];
        unsigned af[8];
#pragma unroll
        for (int v = 0; v < 8; ++v) af[v] = ap[v];
        acc[mt] = wmma_bf16(af, bf, acc[mt]);
      }
    }

    // scale by 1/||corpus_row||, transpose via LDS, update per-lane top-8
    float n2 = ldsRn[lane & 15];
    float scal = 1.f / fmaxf(sqrtf(n2), 1e-12f);
#pragma unroll
    for (int mt = 0; mt < 4; ++mt) {
#pragma unroll
      for (int v = 0; v < 8; ++v) {
        int m = v + 8 * (lane >> 4);
        ldsT[wid * 256 + m * 16 + (lane & 15)] = acc[mt][v] * scal;
      }
      __builtin_amdgcn_wave_barrier();
      int rr = lane & 15;
      int c0 = (lane >> 4) * 8;
#pragma unroll
      for (int j = 0; j < 8; ++j) {
        float vv = ldsT[wid * 256 + rr * 16 + c0 + j];
        top8_insert(topS[mt], topI[mt], vv, t * 16 + c0 + j);
      }
      __builtin_amdgcn_wave_barrier();
    }
  }

  // emit per-(row, half) candidates for this chunk
#pragma unroll
  for (int mt = 0; mt < 4; ++mt) {
    int row = (wid * 4 + mt) * 16 + (lane & 15);
    int half = lane >> 4;
    size_t base = ((size_t)chunk * BSZ + row) * 16 + half * 8;
#pragma unroll
    for (int j = 0; j < 8; ++j)
      cand[base + j] = make_float2(topS[mt][j], __int_as_float(topI[mt][j]));
  }
}

// ================= Kernel 3: final top-8, gather, build shared_in A-frags ==========
__global__ __launch_bounds__(256) void topk_gather_kernel(
    const float2* __restrict__ cand, const float* __restrict__ conf,
    const float* __restrict__ corpus, unsigned* __restrict__ sinF,
    float* __restrict__ d_out) {
  __shared__ float dumpS[2048];
  __shared__ float dumpI[2048];
  __shared__ float row[2176];
  __shared__ float fsc[8];
  __shared__ int   fidx[8];
  const int tid = threadIdx.x, b = blockIdx.x;

  float tS[8]; int tI[8];
#pragma unroll
  for (int j = 0; j < 8; ++j) { tS[j] = -3.4e38f; tI[j] = 0; }
  for (int i = tid; i < NCHUNK * 16; i += 256) {
    int c = i >> 4, rem = i & 15;
    float2 e = cand[((size_t)c * BSZ + b) * 16 + rem];
    top8_insert(tS, tI, e.x, __float_as_int(e.y));
  }
#pragma unroll
  for (int j = 0; j < 8; ++j) { dumpS[tid * 8 + j] = tS[j]; dumpI[tid * 8 + j] = __int_as_float(tI[j]); }
  __syncthreads();
  if (tid == 0) {
    float gS[8]; int gI[8];
#pragma unroll
    for (int j = 0; j < 8; ++j) { gS[j] = -3.4e38f; gI[j] = 0; }
    for (int i = 0; i < 2048; ++i) top8_insert(gS, gI, dumpS[i], __float_as_int(dumpI[i]));
#pragma unroll
    for (int j = 0; j < 8; ++j) { fsc[j] = gS[j]; fidx[j] = gI[j]; }
  }
  __syncthreads();
  if (tid < 8) {
    d_out[O_SC  + b * 8 + tid] = fsc[tid];
    d_out[O_IDX + b * 8 + tid] = (float)fidx[tid];
  }
  // shared_in row = [confounders(128) | retrieved raw embeddings (8*256)]
  if (tid < 128) row[tid] = conf[b * 128 + tid];
  for (int i = tid; i < 2048; i += 256) {
    int j = i >> 8, k = i & 255;
    row[128 + i] = corpus[(size_t)fidx[j] * EMBD + k];
  }
  __syncthreads();
  const int mt = b >> 4, m = b & 15;
  for (int d = tid; d < 1088; d += 256) {
    int k = 2 * d, kk = k & 31, ks = k >> 5;
    int h = (kk >> 3) & 1, v = ((kk & 7) >> 1) + 4 * (kk >> 4);
    int l = m + 16 * h;
    sinF[(((size_t)mt * KS_SIN + ks) * 32 + l) * 8 + v] = pk_bf16(row[k], row[k + 1]);
  }
}

// ================= Kernel 4: generic WMMA GEMM: out = act(Xfrag @ W + bias) =========
__global__ __launch_bounds__(256) void gemm_frag_kernel(
    const unsigned* __restrict__ Xfrag, const float* __restrict__ W,
    const float* __restrict__ bias, float* __restrict__ out,
    int ksc, int N, int relu) {
  __shared__ unsigned ldsW[32 * 256];   // 32 K-steps of a 16-wide W tile (32 KB)
  const int tid = threadIdx.x, lane = tid & 31, wid = tid >> 5;
  const int n0 = blockIdx.x * 16;

  v8f acc[4];
#pragma unroll
  for (int mt = 0; mt < 4; ++mt)
#pragma unroll
    for (int i = 0; i < 8; ++i) acc[mt][i] = 0.f;

  for (int kb = 0; kb < ksc; kb += 32) {
    int kcnt = (ksc - kb < 32) ? (ksc - kb) : 32;
    if (kb + 32 < ksc)  // stream next W block (global_prefetch_b8)
      __builtin_prefetch(&W[(size_t)(kb + 32) * 32 * N + (size_t)tid * 16], 0, 0);
    __syncthreads();
    int nd = kcnt * 256;
    for (int d = tid; d < nd; d += 256) {
      int ks = kb + (d >> 8), rem = d & 255, l = rem >> 3, v = rem & 7;
      int n = n0 + (l & 15);
      int k = ks * 32 + (v >> 2) * 16 + (l >> 4) * 8 + 2 * (v & 3);
      ldsW[d] = pk_bf16(W[(size_t)k * N + n], W[(size_t)(k + 1) * N + n]);
    }
    __syncthreads();
    for (int ks2 = 0; ks2 < kcnt; ++ks2) {
      int ks = kb + ks2;
      unsigned bf[8];
#pragma unroll
      for (int v = 0; v < 8; ++v) bf[v] = ldsW[(ks2 << 8) + lane * 8 + v];
#pragma unroll
      for (int mt = 0; mt < 4; ++mt) {
        const unsigned* ap = &Xfrag[(((size_t)(wid * 4 + mt) * ksc + ks) * 32 + lane) * 8];
        unsigned af[8];
#pragma unroll
        for (int v = 0; v < 8; ++v) af[v] = ap[v];
        acc[mt] = wmma_bf16(af, bf, acc[mt]);
      }
    }
  }
  float bval = bias[n0 + (lane & 15)];
#pragma unroll
  for (int mt = 0; mt < 4; ++mt)
#pragma unroll
    for (int v = 0; v < 8; ++v) {
      int rowm = (wid * 4 + mt) * 16 + 8 * (lane >> 4) + v;
      float r = acc[mt][v] + bval;
      if (relu) r = fmaxf(r, 0.f);
      out[(size_t)rowm * N + n0 + (lane & 15)] = r;
    }
}

// ================= Kernel 5: row LayerNorm -> bf16 A-frags (K=1024) ================
__global__ __launch_bounds__(256) void ln_frag_kernel(
    const float* __restrict__ hpre, const float* __restrict__ g,
    const float* __restrict__ be, unsigned* __restrict__ outF) {
  __shared__ float row[1024];
  __shared__ float red[256];
  const int tid = threadIdx.x, b = blockIdx.x;
  float4 x = reinterpret_cast<const float4*>(hpre + (size_t)b * 1024)[tid];
  red[tid] = x.x + x.y + x.z + x.w;
  __syncthreads();
  for (int off = 128; off > 0; off >>= 1) {
    if (tid < off) red[tid] += red[tid + off];
    __syncthreads();
  }
  float mu = red[0] * (1.f / 1024.f);
  __syncthreads();
  float d0 = x.x - mu, d1 = x.y - mu, d2 = x.z - mu, d3 = x.w - mu;
  red[tid] = d0 * d0 + d1 * d1 + d2 * d2 + d3 * d3;
  __syncthreads();
  for (int off = 128; off > 0; off >>= 1) {
    if (tid < off) red[tid] += red[tid + off];
    __syncthreads();
  }
  float rs = rsqrtf(red[0] * (1.f / 1024.f) + 1e-5f);
  __syncthreads();
  int k0 = tid * 4;
  row[k0 + 0] = d0 * rs * g[k0 + 0] + be[k0 + 0];
  row[k0 + 1] = d1 * rs * g[k0 + 1] + be[k0 + 1];
  row[k0 + 2] = d2 * rs * g[k0 + 2] + be[k0 + 2];
  row[k0 + 3] = d3 * rs * g[k0 + 3] + be[k0 + 3];
  __syncthreads();
  const int mt = b >> 4, m = b & 15;
  for (int d = tid; d < 512; d += 256) {
    int k = 2 * d, kk = k & 31, ks = k >> 5;
    int h = (kk >> 3) & 1, v = ((kk & 7) >> 1) + 4 * (kk >> 4);
    int l = m + 16 * h;
    outF[(((size_t)mt * KS_HID + ks) * 32 + l) * 8 + v] = pk_bf16(row[k], row[k + 1]);
  }
}

// ================= Kernel 6: heads (fp32 VALU; tiny GEMMs) =========================
__global__ __launch_bounds__(256) void heads_kernel(
    const float* __restrict__ treat,
    const float* __restrict__ Wo1, const float* __restrict__ bo1,
    const float* __restrict__ Wo2, const float* __restrict__ bo2,
    const float* __restrict__ Wt1, const float* __restrict__ bt1,
    const float* __restrict__ Wt2, const float* __restrict__ bt2,
    const float* __restrict__ Wg1, const float* __restrict__ bg1,
    const float* __restrict__ Wg2, const float* __restrict__ bg2,
    float* __restrict__ d_out) {
  __shared__ float s[512];
  __shared__ float red[256];
  __shared__ float red2[256];
  const int tid = threadIdx.x, b = blockIdx.x;
  const float* srow = d_out + O_S + (size_t)b * 512;
  s[tid] = srow[tid];
  s[tid + 256] = srow[tid + 256];
  __syncthreads();
  const int tact = (treat[b * 2 + 1] > 0.5f) ? 1 : 0;

  // propensity head
  float a = bt1[tid];
  for (int k = 0; k < 512; ++k) a += s[k] * Wt1[k * 256 + tid];
  a = fmaxf(a, 0.f);
  red[tid]  = a * Wt2[tid * 2 + 0];
  red2[tid] = a * Wt2[tid * 2 + 1];
  __syncthreads();
  for (int off = 128; off > 0; off >>= 1) {
    if (tid < off) { red[tid] += red[tid + off]; red2[tid] += red2[tid + off]; }
    __syncthreads();
  }
  float l0 = red[0] + bt2[0], l1 = red2[0] + bt2[1];
  float mx = fmaxf(l0, l1);
  float e0 = expf(l0 - mx), e1 = expf(l1 - mx), inv = 1.f / (e0 + e1);
  __syncthreads();

  // outcome head: one-hot concat == adding Wo1 row (512+t)
  float base = bo1[tid];
  for (int k = 0; k < 512; ++k) base += s[k] * Wo1[k * 256 + tid];
  float a0 = fmaxf(base + Wo1[512 * 256 + tid], 0.f);
  float a1 = fmaxf(base + Wo1[513 * 256 + tid], 0.f);
  red[tid] = a0 * Wo2[tid];
  red2[tid] = a1 * Wo2[tid];
  __syncthreads();
  for (int off = 128; off > 0; off >>= 1) {
    if (tid < off) { red[tid] += red[tid + off]; red2[tid] += red2[tid + off]; }
    __syncthreads();
  }
  float cf0 = red[0] + bo2[0], cf1 = red2[0] + bo2[0];
  __syncthreads();

  // targeted head (actual treatment only)
  float gb = bg1[tid];
  for (int k = 0; k < 512; ++k) gb += s[k] * Wg1[k * 256 + tid];
  float ga = fmaxf(gb + Wg1[(512 + tact) * 256 + tid], 0.f);
  red[tid] = ga * Wg2[tid];
  __syncthreads();
  for (int off = 128; off > 0; off >>= 1) {
    if (tid < off) red[tid] += red[tid + off];
    __syncthreads();
  }
  float tg = red[0] + bg2[0];

  if (tid == 0) {
    d_out[O_FACT + b] = tact ? cf1 : cf0;
    d_out[O_TARG + b] = tg;
    d_out[O_CF + b * 2 + 0] = cf0;
    d_out[O_CF + b * 2 + 1] = cf1;
    d_out[O_PS + b * 2 + 0] = e0 * inv;
    d_out[O_PS + b * 2 + 1] = e1 * inv;
    d_out[O_PL + b * 2 + 0] = l0;
    d_out[O_PL + b * 2 + 1] = l1;
  }
}

// ================= launch =========================================================
extern "C" void kernel_launch(void* const* d_in, const int* in_sizes, int n_in,
                              void* d_out, int out_size, void* d_ws, size_t ws_size,
                              hipStream_t stream) {
  const float* pf     = (const float*)d_in[0];
  const float* treat  = (const float*)d_in[1];
  const float* conf   = (const float*)d_in[2];
  const float* corpus = (const float*)d_in[3];
  const float* Wpe = (const float*)d_in[4];  const float* bpe = (const float*)d_in[5];
  const float* W1  = (const float*)d_in[6];  const float* b1  = (const float*)d_in[7];
  const float* g1  = (const float*)d_in[8];  const float* be1 = (const float*)d_in[9];
  const float* W2  = (const float*)d_in[10]; const float* b2  = (const float*)d_in[11];
  const float* g2  = (const float*)d_in[12]; const float* be2 = (const float*)d_in[13];
  const float* W3  = (const float*)d_in[14]; const float* b3  = (const float*)d_in[15];
  const float* Wo1 = (const float*)d_in[16]; const float* bo1 = (const float*)d_in[17];
  const float* Wo2 = (const float*)d_in[18]; const float* bo2 = (const float*)d_in[19];
  const float* Wt1 = (const float*)d_in[20]; const float* bt1 = (const float*)d_in[21];
  const float* Wt2 = (const float*)d_in[22]; const float* bt2 = (const float*)d_in[23];
  const float* Wg1 = (const float*)d_in[24]; const float* bg1 = (const float*)d_in[25];
  const float* Wg2 = (const float*)d_in[26]; const float* bg2 = (const float*)d_in[27];
  float* out = (float*)d_out;
  char* ws = (char*)d_ws;

  unsigned* peA  = (unsigned*)(ws + OFF_PEA);
  float2*   cand = (float2*)(ws + OFF_CAND);
  unsigned* sinF = (unsigned*)(ws + OFF_SIN);
  float*    hpre = (float*)(ws + OFF_HPRE);
  unsigned* hF   = (unsigned*)(ws + OFF_HF);

  pe_prep_kernel<<<BSZ, 256, 0, stream>>>(pf, Wpe, bpe, peA);
  sim_topk_kernel<<<NCHUNK, 256, 0, stream>>>(corpus, peA, cand);
  topk_gather_kernel<<<BSZ, 256, 0, stream>>>(cand, conf, corpus, sinF, out);
  gemm_frag_kernel<<<64, 256, 0, stream>>>(sinF, W1, b1, hpre, KS_SIN, 1024, 1);
  ln_frag_kernel<<<BSZ, 256, 0, stream>>>(hpre, g1, be1, hF);
  gemm_frag_kernel<<<64, 256, 0, stream>>>(hF, W2, b2, hpre, KS_HID, 1024, 1);
  ln_frag_kernel<<<BSZ, 256, 0, stream>>>(hpre, g2, be2, hF);
  gemm_frag_kernel<<<32, 256, 0, stream>>>(hF, W3, b3, out + O_S, KS_HID, 512, 0);
  heads_kernel<<<BSZ, 256, 0, stream>>>(treat, Wo1, bo1, Wo2, bo2,
                                        Wt1, bt1, Wt2, bt2, Wg1, bg1, Wg2, bg2, out);
}
